// Adaptive_Softmax_4028679324206
// MI455X (gfx1250) — matrix operand order repaired and checked by reading, not compiled, not run on hardware
//
#include <hip/hip_runtime.h>
#include <math.h>

// ---------- CDNA5 WMMA types ----------
typedef __bf16  bf16_t;
typedef bf16_t  v16bf __attribute__((ext_vector_type(16)));
typedef bf16_t  v4bf  __attribute__((ext_vector_type(4)));
typedef float   v8f   __attribute__((ext_vector_type(8)));

union FragU { uint4 u[2]; v16bf v; };

static __device__ __forceinline__ v8f wmma_bf16(v16bf a, v16bf b, v8f c) {
  // 8 args: (neg_a, A, neg_b, B, c_mod, C, reuse_a, reuse_b)
  return __builtin_amdgcn_wmma_f32_16x16x32_bf16(false, a, false, b, (short)0, c, false, false);
}

// CDNA5 async global->LDS copy (ASYNCcnt-tracked), one b128 per lane.
static __device__ __forceinline__ void async_copy_b128(void* lds_dst, const void* gsrc) {
  unsigned lds_off = (unsigned)(uintptr_t)lds_dst;       // low 32 bits = LDS address
  unsigned long long gaddr = (unsigned long long)(uintptr_t)gsrc;
  asm volatile("global_load_async_to_lds_b128 %0, %1, off"
               :: "v"(lds_off), "v"(gaddr) : "memory");
}
static __device__ __forceinline__ void async_wait() {
  asm volatile("s_wait_asynccnt 0x0" ::: "memory");
}

// online logsumexp helpers
static __device__ __forceinline__ void lse_upd(float v, float& m, float& s) {
  if (v > m) { s = s * __expf(m - v) + 1.0f; m = v; }
  else       { s += __expf(v - m); }
}
static __device__ __forceinline__ void lse_merge(float m2, float s2, float& m, float& s) {
  if (m2 > m) { s = s * __expf(m - m2) + s2; m = m2; }
  else        { s += s2 * __expf(m2 - m); }
}

#define NROWS 2048
#define BM 128
#define BN 128
#define BK 32

// ---------- fp32 -> bf16 elementwise (vectorized, n multiple of 1024) ----------
__global__ __launch_bounds__(256) void cvt_kernel(const float* __restrict__ src,
                                                  bf16_t* __restrict__ dst, int n) {
  int i = (blockIdx.x * 256 + threadIdx.x) * 4;
  if (i < n) {
    float4 v = *(const float4*)(src + i);
    v4bf o;
    o[0] = (bf16_t)v.x; o[1] = (bf16_t)v.y; o[2] = (bf16_t)v.z; o[3] = (bf16_t)v.w;
    *(v4bf*)(dst + i) = o;
  }
}

// ---------- GEMM core macro: bf16 A (global, async DMA) x fp32 B (cvt on the fly) ----------
// 8 waves as 4(M) x 2(N); each wave owns 32 rows x 64 cols = 2x4 WMMA tiles.
// B tile (32k x 128n): kq = tid>>5, n4 = (tid&31)*4; 4 aligned float4 rows each.
// A tile (128m x 32k): two async b128 per thread.
#define GEMM_CORE(A_, B_, K_, V_, colbase_, rowbase_, acc_)                            \
  {                                                                                    \
    const int n4 = lane * 4;                                                           \
    const int kq = w;                                                                  \
    const bool fullTile = (colbase_) + BN <= (V_);                                     \
    const int mA0  = tid >> 2;                 /* rows 0..63  */                       \
    const int mA1  = (tid + 256) >> 2;         /* rows 64..127 */                      \
    const int kcA  = (tid & 3) << 3;                                                   \
    for (int kb = 0; kb < (K_); kb += BK) {                                            \
      /* A tile: 128x32 bf16 via async DMA, two b128 per thread */                     \
      async_copy_b128(&As[mA0][kcA],                                                   \
                      (A_) + (size_t)((rowbase_) + mA0) * (K_) + kb + kcA);            \
      async_copy_b128(&As[mA1][kcA],                                                   \
                      (A_) + (size_t)((rowbase_) + mA1) * (K_) + kb + kcA);            \
      /* B tile: 32x128 fp32 -> bf16, transposed into Bs[n][k] */                      \
      if (fullTile) {                                                                  \
        const float* bp = (B_) + (size_t)(kb + kq) * (V_) + (colbase_) + n4;           \
        float4 r0 = *(const float4*)(bp);                                              \
        float4 r1 = *(const float4*)(bp + (size_t)8  * (V_));                          \
        float4 r2 = *(const float4*)(bp + (size_t)16 * (V_));                          \
        float4 r3 = *(const float4*)(bp + (size_t)24 * (V_));                          \
        Bs[n4+0][kq+ 0]=(bf16_t)r0.x; Bs[n4+1][kq+ 0]=(bf16_t)r0.y;                    \
        Bs[n4+2][kq+ 0]=(bf16_t)r0.z; Bs[n4+3][kq+ 0]=(bf16_t)r0.w;                    \
        Bs[n4+0][kq+ 8]=(bf16_t)r1.x; Bs[n4+1][kq+ 8]=(bf16_t)r1.y;                    \
        Bs[n4+2][kq+ 8]=(bf16_t)r1.z; Bs[n4+3][kq+ 8]=(bf16_t)r1.w;                    \
        Bs[n4+0][kq+16]=(bf16_t)r2.x; Bs[n4+1][kq+16]=(bf16_t)r2.y;                    \
        Bs[n4+2][kq+16]=(bf16_t)r2.z; Bs[n4+3][kq+16]=(bf16_t)r2.w;                    \
        Bs[n4+0][kq+24]=(bf16_t)r3.x; Bs[n4+1][kq+24]=(bf16_t)r3.y;                    \
        Bs[n4+2][kq+24]=(bf16_t)r3.z; Bs[n4+3][kq+24]=(bf16_t)r3.w;                    \
      } else {                                                                         \
        for (int i = 0; i < 4; ++i) {                                                  \
          int k = kq + 8 * i;                                                          \
          for (int j = 0; j < 4; ++j) {                                                \
            int col = (colbase_) + n4 + j;                                             \
            float v = (col < (V_)) ? (B_)[(size_t)(kb + k) * (V_) + col] : 0.0f;       \
            Bs[n4 + j][k] = (bf16_t)v;                                                 \
          }                                                                            \
        }                                                                              \
      }                                                                                \
      async_wait();                                                                    \
      __syncthreads();                                                                 \
      v16bf af[2], bfr[4];                                                             \
      {                                                                                \
        int kca = (lane < 16) ? 0 : 8;   /* A 16-bit layout: lo lanes K0-7/16-23 */    \
        for (int mt = 0; mt < 2; ++mt) {                                               \
          int r = wm * 32 + mt * 16 + (lane & 15);                                     \
          FragU u;                                                                     \
          u.u[0] = *(const uint4*)&As[r][kca];                                         \
          u.u[1] = *(const uint4*)&As[r][kca + 16];                                    \
          af[mt] = u.v;                                                                \
        }                                                                              \
        int kcb = (lane < 16) ? 0 : 8;   /* B 16-bit layout: lo lanes K0-7/16-23 */    \
        for (int nt = 0; nt < 4; ++nt) {                                               \
          int c = wn * 64 + nt * 16 + (lane & 15);                                     \
          FragU u;                                                                     \
          u.u[0] = *(const uint4*)&Bs[c][kcb];                                         \
          u.u[1] = *(const uint4*)&Bs[c][kcb + 16];                                    \
          bfr[nt] = u.v;                                                               \
        }                                                                              \
      }                                                                                \
      for (int mt = 0; mt < 2; ++mt)                                                   \
        for (int nt = 0; nt < 4; ++nt)                                                 \
          acc_[mt][nt] = wmma_bf16(af[mt], bfr[nt], acc_[mt][nt]);                     \
      __syncthreads();                                                                 \
    }                                                                                  \
  }

// ---------- projection GEMM: writes bf16 output ----------
__global__ __launch_bounds__(256) void wmma_gemm_proj(
    const bf16_t* __restrict__ A, const float* __restrict__ B,
    bf16_t* __restrict__ Pb, int K, int V) {
  __shared__ bf16_t As[BM][BK + 8];
  __shared__ bf16_t Bs[BN][BK + 8];
  const int tid = threadIdx.x;
  const int lane = tid & 31;
  const int w = tid >> 5;
  const int wm = w >> 1, wn = w & 1;   // 4 x 2 wave grid
  const int rowbase = blockIdx.x * BM;
  const int colbase = blockIdx.y * BN;

  v8f acc[2][4] = {};
  GEMM_CORE(A, B, K, V, colbase, rowbase, acc)

  for (int mt = 0; mt < 2; ++mt)
    for (int nt = 0; nt < 4; ++nt) {
      int colg = colbase + wn * 64 + nt * 16 + (lane & 15);
      if (colg < V) {
        int rbase = rowbase + wm * 32 + mt * 16 + ((lane >> 4) << 3);
        for (int i = 0; i < 8; ++i)
          Pb[(size_t)(rbase + i) * V + colg] = (bf16_t)acc[mt][nt][i];
      }
    }
}

// ---------- logit GEMM + fused partial logsumexp (two-pass epilogue) ----------
__global__ __launch_bounds__(256) void wmma_gemm_lse(
    const bf16_t* __restrict__ A, const float* __restrict__ B,
    const float* __restrict__ Bias, float2* __restrict__ Part, int K, int V) {
  __shared__ bf16_t As[BM][BK + 8];
  __shared__ bf16_t Bs[BN][BK + 8];
  __shared__ float  Ls[BM][68];   // one 64-col half at a time (+4 pad)
  const int tid = threadIdx.x;
  const int lane = tid & 31;
  const int w = tid >> 5;
  const int wm = w >> 1, wn = w & 1;
  const int rowbase = blockIdx.x * BM;
  const int colbase = blockIdx.y * BN;

  v8f acc[2][4] = {};
  GEMM_CORE(A, B, K, V, colbase, rowbase, acc)

  const int nvalid = min(BN, V - colbase);  // >= 16 for our shapes
  const int row = tid >> 1;
  const int sub = tid & 1;
  float m = -INFINITY, s = 0.0f;

  for (int half = 0; half < 2; ++half) {
    // waves owning this 64-col half spill logits (+bias) to LDS
    if (wn == half) {
      for (int mt = 0; mt < 2; ++mt)
        for (int nt = 0; nt < 4; ++nt) {
          int cl = nt * 16 + (lane & 15);            // local col in [0,64)
          int colg = colbase + half * 64 + cl;
          float bias = (colg < V) ? Bias[colg] : 0.0f;
          int rl = wm * 32 + mt * 16 + ((lane >> 4) << 3);
          for (int i = 0; i < 8; ++i) Ls[rl + i][cl] = acc[mt][nt][i] + bias;
        }
    }
    __syncthreads();
    int lim = min(64, nvalid - half * 64);
    for (int c = sub; c < lim; c += 2) lse_upd(Ls[row][c], m, s);
    __syncthreads();   // Ls reusable for next half
  }
  {  // merge the two sub-threads of each row
    float m2 = __shfl_xor(m, 1, 32);
    float s2 = __shfl_xor(s, 1, 32);
    lse_merge(m2, s2, m, s);
  }
  if (sub == 0) {
    float2 p; p.x = m; p.y = s;
    Part[(size_t)blockIdx.y * NROWS + rowbase + row] = p;
  }
}

// ---------- gathered target logits + cluster logits (wave per row, bf16 projections) ----------
__global__ __launch_bounds__(256) void gather_kernel(
    const bf16_t* __restrict__ p0b, const bf16_t* __restrict__ p1b,
    const bf16_t* __restrict__ p2b, const float* __restrict__ Wl0,
    const float* __restrict__ bl0, const float* __restrict__ Wl1,
    const float* __restrict__ bl1, const float* __restrict__ Wl2,
    const float* __restrict__ bl2, const float* __restrict__ Wc,
    const float* __restrict__ bc, const int* __restrict__ Y,
    float* __restrict__ c01, float* __restrict__ g) {
  int row = blockIdx.x * 8 + (threadIdx.x >> 5);
  int lane = threadIdx.x & 31;
  int y = Y[row];
  int t0 = min(max(y, 0), 9999);
  int t1 = min(max(y - 10000, 0), 29999);
  int t2 = min(max(y - 20000, 0), 51999);
  float c0 = 0.f, c1 = 0.f, g0 = 0.f, g1 = 0.f, g2 = 0.f;
  for (int k = lane; k < 1024; k += 32) {
    float p = (float)p0b[(size_t)row * 1024 + k];
    c0 += p * Wc[k * 2 + 0];
    c1 += p * Wc[k * 2 + 1];
    g0 += p * Wl0[(size_t)k * 10000 + t0];
  }
  for (int k = lane; k < 256; k += 32)
    g1 += (float)p1b[(size_t)row * 256 + k] * Wl1[(size_t)k * 30000 + t1];
  for (int k = lane; k < 64; k += 32)
    g2 += (float)p2b[(size_t)row * 64 + k] * Wl2[(size_t)k * 52000 + t2];
  for (int off = 16; off; off >>= 1) {
    c0 += __shfl_xor(c0, off, 32);
    c1 += __shfl_xor(c1, off, 32);
    g0 += __shfl_xor(g0, off, 32);
    g1 += __shfl_xor(g1, off, 32);
    g2 += __shfl_xor(g2, off, 32);
  }
  if (lane == 0) {
    c01[row * 2 + 0] = c0 + bc[0];
    c01[row * 2 + 1] = c1 + bc[1];
    g[row]             = g0 + bl0[t0];
    g[NROWS + row]     = g1 + bl1[t1];
    g[2 * NROWS + row] = g2 + bl2[t2];
  }
}

// ---------- merge partial (m,s) pairs -> per-row lse (wave per row/cluster) ----------
__global__ __launch_bounds__(256) void reduce_lse_kernel(
    const float2* __restrict__ part0, const float2* __restrict__ part1,
    const float2* __restrict__ part2, const float* __restrict__ c01,
    float* __restrict__ lse, int t0cnt, int t1cnt, int t2cnt) {
  int gid = blockIdx.x * 8 + (threadIdx.x >> 5);
  int lane = threadIdx.x & 31;
  int cluster = gid / NROWS;
  int row = gid % NROWS;
  const float2* part;
  int count;
  if (cluster == 0)      { part = part0; count = t0cnt; }
  else if (cluster == 1) { part = part1; count = t1cnt; }
  else                   { part = part2; count = t2cnt; }
  float m = -INFINITY, s = 0.0f;
  for (int i = lane; i < count; i += 32) {
    float2 p = part[(size_t)i * NROWS + row];
    lse_merge(p.x, p.y, m, s);
  }
  for (int off = 16; off; off >>= 1) {
    float m2 = __shfl_xor(m, off, 32);
    float s2 = __shfl_xor(s, off, 32);
    lse_merge(m2, s2, m, s);
  }
  if (lane == 0) {
    if (cluster == 0) {  // fold the two cluster-logit columns into the head lse
      lse_merge(c01[row * 2 + 0], 1.0f, m, s);
      lse_merge(c01[row * 2 + 1], 1.0f, m, s);
    }
    lse[cluster * NROWS + row] = m + logf(s);
  }
}

// ---------- final masked NLL ----------
__global__ __launch_bounds__(256) void final_nll_kernel(
    const int* __restrict__ Y, const float* __restrict__ g,
    const float* __restrict__ c01, const float* __restrict__ lse,
    float* __restrict__ out) {
  int row = blockIdx.x * 256 + threadIdx.x;
  if (row >= NROWS) return;
  int y = Y[row];
  float lseh = lse[row];
  float r;
  if (y < 10000) {
    r = -(g[row] - lseh);
  } else if (y < 20000) {  // head_lp[:, -1] == cluster col 1
    r = -((c01[row * 2 + 1] - lseh) + (g[NROWS + row] - lse[NROWS + row]));
  } else {                 // head_lp[:, -2] == cluster col 0
    r = -((c01[row * 2 + 0] - lseh) + (g[2 * NROWS + row] - lse[2 * NROWS + row]));
  }
  out[row] = r;
}

extern "C" void kernel_launch(void* const* d_in, const int* in_sizes, int n_in,
                              void* d_out, int out_size, void* d_ws, size_t ws_size,
                              hipStream_t stream) {
  const float* x   = (const float*)d_in[0];
  const int*   y   = (const int*)d_in[1];
  const float* Wp0 = (const float*)d_in[2];
  const float* Wp1 = (const float*)d_in[3];
  const float* Wp2 = (const float*)d_in[4];
  const float* Wl0 = (const float*)d_in[5];
  const float* bl0 = (const float*)d_in[6];
  const float* Wl1 = (const float*)d_in[7];
  const float* bl1 = (const float*)d_in[8];
  const float* Wl2 = (const float*)d_in[9];
  const float* bl2 = (const float*)d_in[10];
  const float* Wc  = (const float*)d_in[11];
  const float* bc  = (const float*)d_in[12];
  float* out = (float*)d_out;

  const int V0 = 10000, V1 = 30000, V2 = 52000;
  const int T0 = (V0 + BN - 1) / BN;  // 79
  const int T1 = (V1 + BN - 1) / BN;  // 235
  const int T2 = (V2 + BN - 1) / BN;  // 407

  size_t off = 0;
  char* base = (char*)d_ws;
  auto take = [&](size_t bytes) -> void* {
    void* p = base + off;
    off = (off + bytes + 255) & ~(size_t)255;
    return p;
  };
  bf16_t* xb  = (bf16_t*)take((size_t)NROWS * 1024 * 2);
  bf16_t* p0b = (bf16_t*)take((size_t)NROWS * 1024 * 2);
  bf16_t* p1b = (bf16_t*)take((size_t)NROWS * 256 * 2);
  bf16_t* p2b = (bf16_t*)take((size_t)NROWS * 64 * 2);
  float2* part0 = (float2*)take((size_t)T0 * NROWS * 8);
  float2* part1 = (float2*)take((size_t)T1 * NROWS * 8);
  float2* part2 = (float2*)take((size_t)T2 * NROWS * 8);
  float* c01 = (float*)take((size_t)NROWS * 2 * 4);
  float* gth = (float*)take((size_t)3 * NROWS * 4);
  float* lse = (float*)take((size_t)3 * NROWS * 4);

  // 1) x -> bf16
  cvt_kernel<<<(NROWS * 1024) / (256 * 4), 256, 0, stream>>>(x, xb, NROWS * 1024);

  // 2) projections (bf16 WMMA): p_i = x @ Wp_i
  wmma_gemm_proj<<<dim3(NROWS / BM, 1024 / BN), 256, 0, stream>>>(xb, Wp0, p0b, 1024, 1024);
  wmma_gemm_proj<<<dim3(NROWS / BM, 256 / BN),  256, 0, stream>>>(xb, Wp1, p1b, 1024, 256);
  wmma_gemm_proj<<<dim3(NROWS / BM, 1),         256, 0, stream>>>(xb, Wp2, p2b, 1024, 64);

  // 3) logit GEMMs fused with partial logsumexp (never materialize full logits)
  wmma_gemm_lse<<<dim3(NROWS / BM, T0), 256, 0, stream>>>(p0b, Wl0, bl0, part0, 1024, V0);
  wmma_gemm_lse<<<dim3(NROWS / BM, T1), 256, 0, stream>>>(p1b, Wl1, bl1, part1, 256, V1);
  wmma_gemm_lse<<<dim3(NROWS / BM, T2), 256, 0, stream>>>(p2b, Wl2, bl2, part2, 64, V2);

  // 4) gathered target logits + cluster logits
  gather_kernel<<<NROWS / 8, 256, 0, stream>>>(p0b, p1b, p2b, Wl0, bl0, Wl1, bl1,
                                               Wl2, bl2, Wc, bc, y, c01, gth);

  // 5) merge partials -> lse per row per cluster
  reduce_lse_kernel<<<(3 * NROWS) / 8, 256, 0, stream>>>(part0, part1, part2, c01, lse,
                                                         T0, T1, T2);

  // 6) final NLL
  final_nll_kernel<<<(NROWS + 255) / 256, 256, 0, stream>>>(y, gth, c01, lse, out);
}
